// Attention_18382460027073
// MI455X (gfx1250) — matrix operand order repaired and checked by reading, not compiled, not run on hardware
//
#include <hip/hip_runtime.h>
#include <stdint.h>

// ---------------------------------------------------------------------------
// CDNA5 types / helpers
// ---------------------------------------------------------------------------
typedef __attribute__((ext_vector_type(16))) __bf16 v16bf;
typedef __attribute__((ext_vector_type(8)))  float  v8f;
typedef int v4i_ __attribute__((vector_size(16)));   // builtin's pointee type

union FragBF { v16bf v; uint4 q[2]; };
union FragAcc { v8f v; float f[8]; };

#ifndef __has_builtin
#define __has_builtin(x) 0
#endif

#if __has_builtin(__builtin_amdgcn_global_load_async_to_lds_b128)
#define HAS_ASYNC 1
#else
#define HAS_ASYNC 0
#endif

// 16-byte global -> LDS copy; async on CDNA5 (ASYNCcnt), sync fallback.
static __device__ inline void async_copy16(const void* g, void* l) {
#if HAS_ASYNC
  __builtin_amdgcn_global_load_async_to_lds_b128(
      (__attribute__((address_space(1))) v4i_*)(uintptr_t)(g),
      (__attribute__((address_space(3))) v4i_*)(uint32_t)(uintptr_t)(l),
      0, 0);
#else
  *(uint4*)l = *(const uint4*)g;
#endif
}

static __device__ inline void wait_async() {
#if HAS_ASYNC
#if __has_builtin(__builtin_amdgcn_s_wait_asynccnt)
  __builtin_amdgcn_s_wait_asynccnt(0);
#else
  asm volatile("s_wait_asynccnt 0" ::: "memory");
#endif
#endif
}

static __device__ inline v8f wmma_bf16(v16bf a, v16bf b, v8f c) {
  // (neg_a, A, neg_b, B, c_mod, C, reuse_a, reuse_b)
  return __builtin_amdgcn_wmma_f32_16x16x32_bf16(false, a, false, b,
                                                 (short)0, c, false, false);
}

// ---------------------------------------------------------------------------
// Problem constants
// ---------------------------------------------------------------------------
#define BATCH   4
#define SEQ     2048
#define HID     2048
#define NHEADS  16
#define HDIM    128
#define MROWS   (BATCH * SEQ)     // 8192
#define QKVN    (3 * HID)         // 6144

// ---------------------------------------------------------------------------
// Kernel 1: fp32 -> bf16 convert (memory bound pre-pass)
// ---------------------------------------------------------------------------
__global__ __launch_bounds__(256) void f32_to_bf16_kernel(const float* __restrict__ in,
                                                          __bf16* __restrict__ out,
                                                          size_t n) {
  size_t i = (size_t)blockIdx.x * blockDim.x + threadIdx.x;
  size_t stride = (size_t)gridDim.x * blockDim.x;
  for (; i < n; i += stride) out[i] = (__bf16)in[i];
}

// ---------------------------------------------------------------------------
// Kernel 2: transpose fp32 [R][C] -> bf16 [C][R]  (weights become [N][K])
// ---------------------------------------------------------------------------
__global__ __launch_bounds__(256) void transpose_f32_to_bf16(const float* __restrict__ in,
                                                             __bf16* __restrict__ out,
                                                             int R, int C) {
  __shared__ float tile[32][33];
  int bx = blockIdx.x * 32;             // C dimension
  int by = blockIdx.y * 32;             // R dimension
  int tx = threadIdx.x & 31;
  int ty = threadIdx.x >> 5;            // 0..7
#pragma unroll
  for (int i = 0; i < 4; ++i)
    tile[ty + i * 8][tx] = in[(size_t)(by + ty + i * 8) * C + bx + tx];
  __syncthreads();
#pragma unroll
  for (int i = 0; i < 4; ++i)
    out[(size_t)(bx + ty + i * 8) * R + by + tx] = (__bf16)tile[tx][ty + i * 8];
}

// ---------------------------------------------------------------------------
// Kernel 3/5: bf16 GEMM  C[M][N] = A[M][K] * Bt[N][K]^T
// 128x128 block tile, 256 threads = 8 waves in 4(M)x2(N) grid, 32x64 per wave.
// K-step 32 (one bf16 WMMA K). LDS tiles staged with async copies.
// ---------------------------------------------------------------------------
#define GLDA 40   // LDS row stride (bf16 elems), multiple of 8 -> 16B aligned
template <bool OUT_F32>
__global__ __launch_bounds__(256) void gemm_bf16_kernel(const __bf16* __restrict__ A,
                                                        const __bf16* __restrict__ Bt,
                                                        void* __restrict__ Cout,
                                                        int M, int N, int K) {
  __shared__ __bf16 lsA[128 * GLDA];
  __shared__ __bf16 lsB[128 * GLDA];

  const int t    = threadIdx.x;
  const int wave = t >> 5, lane = t & 31;
  const int half = lane >> 4, l16 = lane & 15;
  const int wm = wave >> 1;             // 0..3 -> rows wm*32
  const int wn = wave & 1;              // 0..1 -> cols wn*64
  const int m0 = blockIdx.y * 128;
  const int n0 = blockIdx.x * 128;

  FragAcc acc[2][4];
#pragma unroll
  for (int mi = 0; mi < 2; ++mi)
#pragma unroll
    for (int ni = 0; ni < 4; ++ni)
#pragma unroll
      for (int j = 0; j < 8; ++j) acc[mi][ni].f[j] = 0.f;

  const int ksteps = K >> 5;
  for (int ks = 0; ks < ksteps; ++ks) {
    const int k0 = ks << 5;
    // Stage A/B tiles: 128 rows x 32 bf16 each = 512 16B chunks per tile.
#pragma unroll
    for (int i = 0; i < 2; ++i) {
      int c   = t + i * 256;            // 0..511
      int row = c >> 2;
      int ko  = (c & 3) * 8;
      async_copy16(A  + (size_t)(m0 + row) * K + k0 + ko, &lsA[row * GLDA + ko]);
      async_copy16(Bt + (size_t)(n0 + row) * K + k0 + ko, &lsB[row * GLDA + ko]);
    }
    wait_async();
    __syncthreads();

    FragBF a[2], b[4];
    const int klo = half * 8;
#pragma unroll
    for (int mi = 0; mi < 2; ++mi) {
      int row = wm * 32 + mi * 16 + l16;
      a[mi].q[0] = *(const uint4*)&lsA[row * GLDA + klo];
      a[mi].q[1] = *(const uint4*)&lsA[row * GLDA + 16 + klo];
    }
    const int koB = half * 8;
#pragma unroll
    for (int ni = 0; ni < 4; ++ni) {
      int row = wn * 64 + ni * 16 + l16;
      b[ni].q[0] = *(const uint4*)&lsB[row * GLDA + koB];
      b[ni].q[1] = *(const uint4*)&lsB[row * GLDA + koB + 16];
    }
#pragma unroll
    for (int mi = 0; mi < 2; ++mi)
#pragma unroll
      for (int ni = 0; ni < 4; ++ni)
        acc[mi][ni].v = wmma_bf16(a[mi].v, b[ni].v, acc[mi][ni].v);
    __syncthreads();
  }

  // Epilogue: C layout -> lane holds col l16, rows g + 8*half.
#pragma unroll
  for (int mi = 0; mi < 2; ++mi)
#pragma unroll
    for (int ni = 0; ni < 4; ++ni)
#pragma unroll
      for (int g = 0; g < 8; ++g) {
        int r = m0 + wm * 32 + mi * 16 + g + half * 8;
        int c = n0 + wn * 64 + ni * 16 + l16;
        if (OUT_F32) ((float*)Cout)[(size_t)r * N + c] = acc[mi][ni].f[g];
        else ((__bf16*)Cout)[(size_t)r * N + c] = (__bf16)acc[mi][ni].f[g];
      }
}

// ---------------------------------------------------------------------------
// Kernel 4: flash attention over the faithful-`.view` scrambled q/k/v.
// q[b,h,t,d] = qkv[b, 128*h + t/16, (t%16)*384 + d] (k:+128, v:+256)
// Block: 128 query rows of one (b,h); 8 waves x 16 rows. Key chunks of 64.
// ---------------------------------------------------------------------------
#define TK   64
#define KLD  136   // lsK row stride (bf16), mult of 8
#define VLD  72    // lsV / lsP row stride
__global__ __launch_bounds__(256) void flash_attn_kernel(const __bf16* __restrict__ qkv,
                                                         __bf16* __restrict__ attn) {
  __shared__ __bf16 lsK[TK * KLD];        // [key][d]
  __shared__ __bf16 lsV[HDIM * VLD];      // [d][key] (transposed)
  __shared__ __bf16 lsP[8][16 * VLD];     // per-wave P tile [qrow][key]

  const int t    = threadIdx.x;
  const int wave = t >> 5, lane = t & 31;
  const int half = lane >> 4, l16 = lane & 15;
  const int bh = blockIdx.y;
  const int b = bh >> 4, h = bh & 15;
  const int q0 = blockIdx.x * 128;

  const size_t base = (size_t)b * SEQ * QKVN;
  auto qkvoff = [&](int tt, int dd) -> size_t {
    return base + (size_t)(128 * h + (tt >> 4)) * QKVN + (size_t)(tt & 15) * 384 + dd;
  };

  // Q fragments in registers for the whole kernel (A-matrix layout).
  FragBF qf[4];
  {
    const int qr  = q0 + wave * 16 + l16;
    const int klo = half * 8;
#pragma unroll
    for (int kd = 0; kd < 4; ++kd) {
      int db = kd * 32;
      qf[kd].q[0] = *(const uint4*)(qkv + qkvoff(qr, db + klo));
      qf[kd].q[1] = *(const uint4*)(qkv + qkvoff(qr, db + 16 + klo));
    }
  }

  float m_[8], l_[8];
  FragAcc o[8];
#pragma unroll
  for (int g = 0; g < 8; ++g) { m_[g] = -3.0e38f; l_[g] = 0.f; }
#pragma unroll
  for (int dt = 0; dt < 8; ++dt)
#pragma unroll
    for (int g = 0; g < 8; ++g) o[dt].f[g] = 0.f;

  const float SC = 0.08838834764f * 1.44269504f;   // (1/sqrt(128)) * log2(e)

  for (int kc = 0; kc < SEQ / TK; ++kc) {
    const int tb = kc * TK;
    // K chunk -> LDS (async gather; scramble folded into addresses).
#pragma unroll
    for (int i = 0; i < 4; ++i) {
      int c = t + i * 256;              // 0..1023
      int row = c >> 4, doff = (c & 15) * 8;
      async_copy16(qkv + qkvoff(tb + row, 128 + doff), &lsK[row * KLD + doff]);
    }
    // V chunk -> LDS transposed ([d][key]) via per-element ds stores.
#pragma unroll
    for (int i = 0; i < 4; ++i) {
      int c = t + i * 256;
      int row = c >> 4, doff = (c & 15) * 8;
      uint4 u = *(const uint4*)(qkv + qkvoff(tb + row, 256 + doff));
      const __bf16* e = (const __bf16*)&u;
#pragma unroll
      for (int j = 0; j < 8; ++j) lsV[(doff + j) * VLD + row] = e[j];
    }
    wait_async();
    __syncthreads();

    // S = Q * K^T : 4 key tiles x 4 d-steps.
    FragAcc s[4];
#pragma unroll
    for (int nt = 0; nt < 4; ++nt) {
#pragma unroll
      for (int j = 0; j < 8; ++j) s[nt].f[j] = 0.f;
#pragma unroll
      for (int kd = 0; kd < 4; ++kd) {
        FragBF kb;
        int krow = nt * 16 + l16;
        int dof  = kd * 32 + half * 8;
        kb.q[0] = *(const uint4*)&lsK[krow * KLD + dof];
        kb.q[1] = *(const uint4*)&lsK[krow * KLD + dof + 16];
        s[nt].v = wmma_bf16(qf[kd].v, kb.v, s[nt].v);
      }
    }

    // Online softmax per row (row g+8*half lives in a 16-lane group).
#pragma unroll
    for (int g = 0; g < 8; ++g) {
      float x = s[0].f[g];
#pragma unroll
      for (int nt = 1; nt < 4; ++nt) x = fmaxf(x, s[nt].f[g]);
#pragma unroll
      for (int d = 1; d < 16; d <<= 1) x = fmaxf(x, __shfl_xor(x, d));
      float mn = fmaxf(m_[g], x);
      float alpha = exp2f((m_[g] - mn) * SC);
      m_[g] = mn;
      float rs = 0.f;
#pragma unroll
      for (int nt = 0; nt < 4; ++nt) {
        float p = exp2f((s[nt].f[g] - mn) * SC);
        s[nt].f[g] = p;
        rs += p;
      }
#pragma unroll
      for (int d = 1; d < 16; d <<= 1) rs += __shfl_xor(rs, d);
      l_[g] = l_[g] * alpha + rs;
#pragma unroll
      for (int dt = 0; dt < 8; ++dt) o[dt].f[g] *= alpha;
      int prow = g + half * 8;
#pragma unroll
      for (int nt = 0; nt < 4; ++nt)
        lsP[wave][prow * VLD + nt * 16 + l16] = (__bf16)s[nt].f[g];
    }
    // lsP is per-wave and LDS ops are in-order within a wave: no barrier.

    // O += P * V : 8 d tiles x 2 key-steps.
#pragma unroll
    for (int dt = 0; dt < 8; ++dt)
#pragma unroll
      for (int kk = 0; kk < 2; ++kk) {
        FragBF pa, vb;
        int pko = kk * 32 + half * 8;
        pa.q[0] = *(const uint4*)&lsP[wave][l16 * VLD + pko];
        pa.q[1] = *(const uint4*)&lsP[wave][l16 * VLD + pko + 16];
        int vrow = dt * 16 + l16;
        int vko  = kk * 32 + half * 8;
        vb.q[0] = *(const uint4*)&lsV[vrow * VLD + vko];
        vb.q[1] = *(const uint4*)&lsV[vrow * VLD + vko + 16];
        o[dt].v = wmma_bf16(pa.v, vb.v, o[dt].v);
      }
    __syncthreads();   // protect lsK/lsV before next chunk's copies
  }

  // Normalize and store with the faithful-.reshape scramble:
  // attn[(b*2048 + 128*h + t/16)*2048 + (t%16)*128 + d]
  float rinv[8];
#pragma unroll
  for (int g = 0; g < 8; ++g) rinv[g] = 1.f / l_[g];
  const size_t obase = (size_t)b * SEQ * HID;
#pragma unroll
  for (int dt = 0; dt < 8; ++dt)
#pragma unroll
    for (int g = 0; g < 8; ++g) {
      int tt = q0 + wave * 16 + g + half * 8;
      int dd = dt * 16 + l16;
      size_t off = obase + (size_t)(128 * h + (tt >> 4)) * HID
                 + (size_t)(tt & 15) * HDIM + dd;
      attn[off] = (__bf16)(o[dt].f[g] * rinv[g]);
    }
}

// ---------------------------------------------------------------------------
// Launch
// ---------------------------------------------------------------------------
extern "C" void kernel_launch(void* const* d_in, const int* in_sizes, int n_in,
                              void* d_out, int out_size, void* d_ws, size_t ws_size,
                              hipStream_t stream) {
  (void)in_sizes; (void)n_in; (void)out_size; (void)ws_size;
  const float* hidden = (const float*)d_in[0];   // (4, 2048, 2048)
  const float* w_qkv  = (const float*)d_in[1];   // (2048, 6144)
  const float* w_o    = (const float*)d_in[2];   // (2048, 2048)

  char* ws = (char*)d_ws;
  __bf16* hb    = (__bf16*)ws; ws += (size_t)MROWS * HID * 2;        // 33.5 MB
  __bf16* wqkvT = (__bf16*)ws; ws += (size_t)QKVN * HID * 2;         // 25.2 MB
  __bf16* woT   = (__bf16*)ws; ws += (size_t)HID * HID * 2;          //  8.4 MB
  __bf16* qkv   = (__bf16*)ws; ws += (size_t)MROWS * QKVN * 2;       // 100.7 MB
  __bf16* attn  = (__bf16*)ws;                                       // 33.5 MB

  // 1) convert activations to bf16
  f32_to_bf16_kernel<<<2048, 256, 0, stream>>>(hidden, hb, (size_t)MROWS * HID);
  // 2) transpose+convert weights to [N][K] bf16
  transpose_f32_to_bf16<<<dim3(QKVN / 32, HID / 32), 256, 0, stream>>>(w_qkv, wqkvT, HID, QKVN);
  transpose_f32_to_bf16<<<dim3(HID / 32, HID / 32), 256, 0, stream>>>(w_o, woT, HID, HID);
  // 3) qkv = hidden @ w_qkv  (bf16 out)
  gemm_bf16_kernel<false><<<dim3(QKVN / 128, MROWS / 128), 256, 0, stream>>>(
      hb, wqkvT, (void*)qkv, MROWS, QKVN, HID);
  // 4) fused flash attention (scrambled q/k/v gather + scrambled output scatter)
  flash_attn_kernel<<<dim3(SEQ / 128, BATCH * NHEADS), 256, 0, stream>>>(qkv, attn);
  // 5) out = attn @ w_o  (fp32 out)
  gemm_bf16_kernel<true><<<dim3(HID / 128, MROWS / 128), 256, 0, stream>>>(
      attn, woT, d_out, MROWS, HID, HID);
}
